// MultiHeadAttention_34144990003301
// MI455X (gfx1250) — matrix operand order repaired and checked by reading, not compiled, not run on hardware
//
#include <hip/hip_runtime.h>
#include <math.h>

#define BS 32
#define HEADS 16
#define FN 124
#define SL 256
#define KN 64
#define HID 5000
#define FEAT 16384   // SL*KN
#define ROWS 512     // BS*HEADS
#define EPS 1e-5f

typedef __attribute__((ext_vector_type(16))) __bf16 v16bf;
typedef __attribute__((ext_vector_type(8)))  float  v8f;
typedef __attribute__((ext_vector_type(4)))  unsigned int u32x4;
typedef __attribute__((ext_vector_type(8)))  int i32x8;
typedef __attribute__((ext_vector_type(4)))  int i32x4;

#define WMMA_BF16(A,B,C) \
  __builtin_amdgcn_wmma_f32_16x16x32_bf16(false,(A),false,(B),(short)0,(C),false,false)

#if defined(__has_builtin)
#  if __has_builtin(__builtin_amdgcn_tensor_load_to_lds) && \
      __has_builtin(__builtin_amdgcn_s_wait_tensorcnt)
#    define HAVE_TDM 1
#  endif
#endif
#ifndef HAVE_TDM
#  define HAVE_TDM 0
#endif

// A-matrix (16-bit, 16x32) element->K mapping used here (contiguous; B operands use the same order):
// lane L: M = L%16, half = L/16; vector element i holds K below.
__device__ __forceinline__ int a_kidx(int i, int half){
  return 16*half + i;
}

__device__ __forceinline__ float red_max16(float v){
  #pragma unroll
  for (int m = 1; m < 16; m <<= 1) v = fmaxf(v, __shfl_xor(v, m, 16));
  return v;
}
__device__ __forceinline__ float red_sum16(float v){
  #pragma unroll
  for (int m = 1; m < 16; m <<= 1) v += __shfl_xor(v, m, 16);
  return v;
}

#if HAVE_TDM
// TDM: DMA a [rows<=64] x [128 float] tile of W1 (row stride 16384 floats) into LDS.
// D# per CDNA5 ISA ch.8: group0 = {count, lds_addr, global_addr, type=2},
// group1 = {data_size=4B, tensor_dim0=128, tensor_dim1=rows, tile=128x64, stride0=16384}.
__device__ __forceinline__ void tdm_load_w1(const float* gptr, unsigned lds_off, int rows){
  unsigned long long ga = (unsigned long long)(size_t)gptr;
  u32x4 g0;
  g0[0] = 1u;                                            // count=1 (user descriptor)
  g0[1] = lds_off;                                       // lds_addr (bytes)
  g0[2] = (unsigned)(ga & 0xffffffffu);                  // global_addr[31:0]
  g0[3] = (unsigned)((ga >> 32) & 0x01ffffffu) | (2u << 30); // addr[56:32] | type=2
  i32x8 g1;
  g1[0] = (int)(2u << 16);                               // data_size=2 (4 bytes)
  g1[1] = (int)(128u << 16);                             // tensor_dim0[15:0]=128
  g1[2] = (int)(((unsigned)rows) << 16);                 // dim0 hi=0 | tensor_dim1 lo=rows
  g1[3] = (int)(128u << 16);                             // dim1 hi=0 | tile_dim0=128
  g1[4] = 64;                                            // tile_dim1=64 | tile_dim2=0
  g1[5] = 16384;                                         // tensor_dim0_stride lo32
  g1[6] = 0;                                             // stride hi | dim1_stride lo
  g1[7] = 0;
  i32x4 z4 = {0, 0, 0, 0};
#if __clang_major__ >= 23
  i32x8 z8 = {0, 0, 0, 0, 0, 0, 0, 0};
  __builtin_amdgcn_tensor_load_to_lds(g0, g1, z4, z4, z8, 0);
#else
  __builtin_amdgcn_tensor_load_to_lds(g0, g1, z4, z4, 0);
#endif
}
#endif

// ---------------- stage 0: zero accumulators ----------------
__global__ void k_init(float* __restrict__ rs, float* __restrict__ ostats){
  int i = threadIdx.x;
  for (int j = i; j < 744; j += 256) rs[j] = 0.f;
  if (i < 32) ostats[i] = 0.f;
}

// ---------------- stage 1: projection + row stats ----------------
__global__ void k_proj(const float* __restrict__ x, const float* __restrict__ W,
                       const float* __restrict__ bias, float* __restrict__ P,
                       float* __restrict__ rowstats){
  int g = blockIdx.x, b = blockIdx.y, s = threadIdx.x;
  const float* xb = x + (size_t)b*FN*SL;
  const float* wr = W + g*FN;            // uniform -> scalar loads
  float acc = bias[g];
  #pragma unroll 4
  for (int f = 0; f < FN; ++f) acc = fmaf(wr[f], xb[f*SL + s], acc);
  P[((size_t)b*FN + g)*SL + s] = acc;
  float v1 = acc, v2 = acc*acc;
  #pragma unroll
  for (int m = 1; m < 32; m <<= 1){ v1 += __shfl_xor(v1, m, 32); v2 += __shfl_xor(v2, m, 32); }
  __shared__ float s1[8], s2[8];
  int lane = s & 31, w = s >> 5;
  if (lane == 0){ s1[w] = v1; s2[w] = v2; }
  __syncthreads();
  if (s == 0){
    float a = 0.f, c = 0.f;
    #pragma unroll
    for (int i = 0; i < 8; ++i){ a += s1[i]; c += s2[i]; }
    atomicAdd(&rowstats[2*g],   a);
    atomicAdd(&rowstats[2*g+1], c);
  }
}

// ---------------- stage 2: per-head BN params (windowed) ----------------
__global__ void k_bnhead(const float* gq, const float* gk, const float* gv,
                         const float* beq, const float* bek, const float* bev,
                         const float* __restrict__ rowstats, float* __restrict__ hp){
  int idx = threadIdx.x;
  if (idx >= 48) return;
  int mat = idx / 16, h = idx % 16;
  const float* rs = rowstats + mat*248;
  float s = 0.f, s2 = 0.f;
  for (int k = 0; k < KN; ++k){ int g = h*4 + k; s += rs[2*g]; s2 += rs[2*g+1]; }
  float invN = 1.f / (64.f * (float)(BS*SL));
  float mean = s*invN, var = s2*invN - mean*mean;
  float gam = (mat == 0 ? gq : (mat == 1 ? gk : gv))[h];
  float bet = (mat == 0 ? beq : (mat == 1 ? bek : bev))[h];
  float scl = gam * rsqrtf(var + EPS);
  hp[idx*2]   = scl;
  hp[idx*2+1] = bet - mean*scl;
}

// ---------------- stage 3: flash attention per (b,h), WMMA bf16 ----------------
__global__ void __launch_bounds__(256) k_attn(
    const float* __restrict__ PQ, const float* __restrict__ PK,
    const float* __restrict__ PV, const float* __restrict__ hp,
    float* __restrict__ O, float* __restrict__ ostats){
  extern __shared__ char smem[];
  __bf16* Kn    = (__bf16*)smem;     // [256][64] normalized K
  __bf16* Vn    = Kn + SL*KN;        // [256][64] normalized V
  __bf16* stage = Vn + SL*KN;        // 8 waves x [16][32] P relayout scratch
  int h = blockIdx.x, b = blockIdx.y;
  int tid = threadIdx.x, lane = tid & 31, wave = tid >> 5;
  int lm = lane & 15, half = lane >> 4, h4 = h*4;
  float sq = hp[h*2],        tq = hp[h*2 + 1];
  float sk = hp[32 + h*2],   tk = hp[32 + h*2 + 1];
  float sv = hp[64 + h*2],   tv = hp[64 + h*2 + 1];
  const float* qpb = PQ + (size_t)b*FN*SL;
  const float* kpb = PK + (size_t)b*FN*SL;
  const float* vpb = PV + (size_t)b*FN*SL;
  for (int idx = tid; idx < SL*KN; idx += 256){
    int kk = idx >> 8, s = idx & 255;                       // coalesced in s
    Kn[s*KN + kk] = (__bf16)(kpb[(h4+kk)*SL + s]*sk + tk);
    Vn[s*KN + kk] = (__bf16)(vpb[(h4+kk)*SL + s]*sv + tv);
  }
  __syncthreads();
  __bf16* stg = stage + wave*512;
  float* Og = O + ((size_t)(b*HEADS + h))*SL*KN;
  float bsum = 0.f, bsq = 0.f;
  v8f vzero = {};
  for (int it = 0; it < 2; ++it){
    int i0 = (wave*2 + it)*16;                              // 16-row tile
    v16bf aq0, aq1;                                         // Q A-frags (K=0..31, 32..63)
    #pragma unroll
    for (int i = 0; i < 16; ++i){
      int K = a_kidx(i, half);
      aq0[i] = (__bf16)(qpb[(h4 + K)*SL      + i0 + lm]*sq + tq);
      aq1[i] = (__bf16)(qpb[(h4 + 32 + K)*SL + i0 + lm]*sq + tq);
    }
    float m_i[8], l_i[8];
    #pragma unroll
    for (int r = 0; r < 8; ++r){ m_i[r] = -1e30f; l_i[r] = 0.f; }
    v8f accv[4];
    #pragma unroll
    for (int nt = 0; nt < 4; ++nt) accv[nt] = vzero;
    for (int j0 = 0; j0 < SL; j0 += 32){
      v8f S0 = vzero, S1 = vzero;
      {
        v16bf bk0, bk1, bk2, bk3;     // B layout: N = lane%16 (key), K = i + 16*half (feat)
        const __bf16* kr  = Kn + (j0 + lm)*KN      + 16*half;
        const __bf16* kr2 = Kn + (j0 + 16 + lm)*KN + 16*half;
        #pragma unroll
        for (int i = 0; i < 16; ++i){
          bk0[i] = kr[i];  bk1[i] = kr[32 + i];
          bk2[i] = kr2[i]; bk3[i] = kr2[32 + i];
        }
        S0 = WMMA_BF16(aq0, bk0, S0);  S0 = WMMA_BF16(aq1, bk1, S0);
        S1 = WMMA_BF16(aq0, bk2, S1);  S1 = WMMA_BF16(aq1, bk3, S1);
      }
      #pragma unroll
      for (int r = 0; r < 8; ++r){
        float s0 = S0[r]*0.125f, s1 = S1[r]*0.125f;        // 1/sqrt(KN)
        float tm = red_max16(fmaxf(s0, s1));
        float mn = fmaxf(m_i[r], tm);
        float al = __expf(m_i[r] - mn);
        m_i[r] = mn;
        float p0 = __expf(s0 - mn), p1 = __expf(s1 - mn);
        l_i[r] = l_i[r]*al + red_sum16(p0 + p1);
        stg[(r + 8*half)*32 + lm]      = (__bf16)p0;       // C-layout -> LDS
        stg[(r + 8*half)*32 + 16 + lm] = (__bf16)p1;
        #pragma unroll
        for (int nt = 0; nt < 4; ++nt) accv[nt][r] *= al;
      }
      __asm__ volatile("" ::: "memory");   // per-wave DS ops are in-order; block reordering
      v16bf ap;                            // reload P in A layout
      #pragma unroll
      for (int i = 0; i < 16; ++i) ap[i] = stg[lm*32 + a_kidx(i, half)];
      #pragma unroll
      for (int nt = 0; nt < 4; ++nt){
        v16bf bvv;                         // B = V rows j0..j0+31, cols nt*16..+15
        #pragma unroll
        for (int i = 0; i < 16; ++i) bvv[i] = Vn[(j0 + i + 16*half)*KN + nt*16 + lm];
        accv[nt] = WMMA_BF16(ap, bvv, accv[nt]);
      }
      __asm__ volatile("" ::: "memory");
    }
    #pragma unroll
    for (int r = 0; r < 8; ++r){
      float inv = 1.f / l_i[r];
      int row = i0 + r + 8*half;
      #pragma unroll
      for (int nt = 0; nt < 4; ++nt){
        float vv = accv[nt][r]*inv;
        Og[(size_t)row*KN + nt*16 + lm] = vv;
        bsum += vv; bsq += vv*vv;
      }
    }
  }
  #pragma unroll
  for (int m = 1; m < 32; m <<= 1){ bsum += __shfl_xor(bsum, m, 32); bsq += __shfl_xor(bsq, m, 32); }
  if (lane == 0){ atomicAdd(&ostats[2*h], bsum); atomicAdd(&ostats[2*h+1], bsq); }
}

// ---------------- stage 4: finalize O BN params ----------------
__global__ void k_bn2(const float* g1, const float* be1,
                      const float* __restrict__ ostats, float* __restrict__ op){
  int h = threadIdx.x;
  if (h >= 16) return;
  float invN = 1.f / (float)(BS*SL*KN);
  float mean = ostats[2*h]*invN;
  float var  = ostats[2*h+1]*invN - mean*mean;
  float scl  = g1[h]*rsqrtf(var + EPS);
  op[2*h] = scl; op[2*h+1] = be1[h] - mean*scl;
}

// ---------------- stage 5: normalize O -> bf16 X ----------------
__global__ void k_normo(const float* __restrict__ O, const float* __restrict__ op,
                        __bf16* __restrict__ X){
  int idx = blockIdx.x*256 + threadIdx.x;        // exactly ROWS*FEAT elements
  int hh = (idx >> 14) & 15;                     // row r = idx/16384, h = r%16
  X[idx] = (__bf16)(O[idx]*op[2*hh] + op[2*hh+1]);
}

// ---------------- stage 6: H1 = leaky(X @ W1^T + b1) -- TDM-fed WMMA GEMM ----------------
__global__ void __launch_bounds__(256) k_gemm1(
    const __bf16* __restrict__ X, const float* __restrict__ W1,
    const float* __restrict__ b1, float* __restrict__ H1){
  int tid = threadIdx.x, lane = tid & 31, wave = tid >> 5;
  int lm = lane & 15, half = lane >> 4;
  int m0 = blockIdx.y*128 + (wave >> 1)*32;      // 2 M-tiles per wave
  int o0 = blockIdx.x*64  + (wave & 1)*32;       // 2 N-tiles per wave
  v8f vzero = {};
  v8f acc[2][2];
  acc[0][0] = vzero; acc[0][1] = vzero; acc[1][0] = vzero; acc[1][1] = vzero;
  const __bf16* xb0 = X + (size_t)(m0 + lm)*FEAT      + 16*half;
  const __bf16* xb1 = X + (size_t)(m0 + 16 + lm)*FEAT + 16*half;
#if HAVE_TDM
  extern __shared__ float wtile[];               // double-buffered 64x128 fp32 W1 tiles
  int o_blk = blockIdx.x*64;
  int rows = HID - o_blk; if (rows > 64) rows = 64;   // edge tiles: TDM OOB reads -> 0
  unsigned ldsA = (unsigned)(size_t)(&wtile[0]);
  unsigned ldsB = (unsigned)(size_t)(&wtile[64*128]);
  int wofs = (wave & 1)*32;                      // wave's o-offset inside the 64-row slab
  if (wave == 0) tdm_load_w1(W1 + (size_t)o_blk*FEAT, ldsA, rows);
  for (int c = 0; c < FEAT/128; ++c){
    __syncthreads();                             // buffer (c+1)&1 no longer being read
    if (wave == 0){
      if (c + 1 < FEAT/128){
        tdm_load_w1(W1 + (size_t)o_blk*FEAT + (c + 1)*128, (c & 1) ? ldsA : ldsB, rows);
        __builtin_amdgcn_s_wait_tensorcnt(1);    // chunk c landed (TDM ops in-order)
      } else {
        __builtin_amdgcn_s_wait_tensorcnt(0);
      }
    }
    __syncthreads();                             // chunk c visible to all waves
    const float* wt = (c & 1) ? (wtile + 64*128) : wtile;
    int fc = c*128;
    __builtin_prefetch(xb0 + fc + 4096, 0, 1);   // stream X ahead (global_prefetch_b8)
    #pragma unroll
    for (int ks = 0; ks < 4; ++ks){
      int f0 = fc + ks*32;
      union { uint4 u[2]; v16bf v; } ua0, ua1;   // A runs match A-layout element order
      ua0.u[0] = *(const uint4*)(xb0 + f0); ua0.u[1] = *(const uint4*)(xb0 + f0 + 8);
      ua1.u[0] = *(const uint4*)(xb1 + f0); ua1.u[1] = *(const uint4*)(xb1 + f0 + 8);
      v16bf bb[2];
      #pragma unroll
      for (int nt = 0; nt < 2; ++nt){            // B from LDS tile: K = i + 16*half
        const float* wr = wt + (wofs + nt*16 + lm)*128 + ks*32 + 16*half;
        float4 w0 = *(const float4*)(wr);
        float4 w1 = *(const float4*)(wr + 4);
        float4 w2 = *(const float4*)(wr + 8);
        float4 w3 = *(const float4*)(wr + 12);
        v16bf t;
        t[0]=(__bf16)w0.x; t[1]=(__bf16)w0.y; t[2]=(__bf16)w0.z; t[3]=(__bf16)w0.w;
        t[4]=(__bf16)w1.x; t[5]=(__bf16)w1.y; t[6]=(__bf16)w1.z; t[7]=(__bf16)w1.w;
        t[8]=(__bf16)w2.x; t[9]=(__bf16)w2.y; t[10]=(__bf16)w2.z; t[11]=(__bf16)w2.w;
        t[12]=(__bf16)w3.x; t[13]=(__bf16)w3.y; t[14]=(__bf16)w3.z; t[15]=(__bf16)w3.w;
        bb[nt] = t;
      }
      acc[0][0] = WMMA_BF16(ua0.v, bb[0], acc[0][0]);
      acc[0][1] = WMMA_BF16(ua0.v, bb[1], acc[0][1]);
      acc[1][0] = WMMA_BF16(ua1.v, bb[0], acc[1][0]);
      acc[1][1] = WMMA_BF16(ua1.v, bb[1], acc[1][1]);
    }
  }
#else
  int o_a = o0 + lm;      if (o_a > HID-1) o_a = HID-1;
  int o_b = o0 + 16 + lm; if (o_b > HID-1) o_b = HID-1;
  const float* wb0 = W1 + (size_t)o_a*FEAT + 16*half;
  const float* wb1 = W1 + (size_t)o_b*FEAT + 16*half;
  #pragma unroll 1
  for (int f0 = 0; f0 < FEAT; f0 += 32){
    union { uint4 u[2]; v16bf v; } ua0, ua1;
    ua0.u[0] = *(const uint4*)(xb0 + f0); ua0.u[1] = *(const uint4*)(xb0 + f0 + 8);
    ua1.u[0] = *(const uint4*)(xb1 + f0); ua1.u[1] = *(const uint4*)(xb1 + f0 + 8);
    v16bf bb[2];
    #pragma unroll
    for (int nt = 0; nt < 2; ++nt){
      const float* wr = (nt ? wb1 : wb0) + f0;
      __builtin_prefetch(wr + 2048, 0, 1);
      float4 w0 = *(const float4*)(wr);
      float4 w1 = *(const float4*)(wr + 4);
      float4 w2 = *(const float4*)(wr + 8);
      float4 w3 = *(const float4*)(wr + 12);
      v16bf t;
      t[0]=(__bf16)w0.x; t[1]=(__bf16)w0.y; t[2]=(__bf16)w0.z; t[3]=(__bf16)w0.w;
      t[4]=(__bf16)w1.x; t[5]=(__bf16)w1.y; t[6]=(__bf16)w1.z; t[7]=(__bf16)w1.w;
      t[8]=(__bf16)w2.x; t[9]=(__bf16)w2.y; t[10]=(__bf16)w2.z; t[11]=(__bf16)w2.w;
      t[12]=(__bf16)w3.x; t[13]=(__bf16)w3.y; t[14]=(__bf16)w3.z; t[15]=(__bf16)w3.w;
      bb[nt] = t;
    }
    acc[0][0] = WMMA_BF16(ua0.v, bb[0], acc[0][0]);
    acc[0][1] = WMMA_BF16(ua0.v, bb[1], acc[0][1]);
    acc[1][0] = WMMA_BF16(ua1.v, bb[0], acc[1][0]);
    acc[1][1] = WMMA_BF16(ua1.v, bb[1], acc[1][1]);
  }
#endif
  #pragma unroll
  for (int mt = 0; mt < 2; ++mt){
    #pragma unroll
    for (int nt = 0; nt < 2; ++nt){
      int o = o0 + nt*16 + lm;
      if (o < HID){
        float bo = b1[o];
        #pragma unroll
        for (int r = 0; r < 8; ++r){
          int row = m0 + mt*16 + r + 8*half;     // C-layout row
          float vv = acc[mt][nt][r] + bo;
          vv = (vv >= 0.f) ? vv : 0.01f*vv;      // leaky relu
          H1[(size_t)row*HID + o] = vv;
        }
      }
    }
  }
}

// ---------------- stage 7: out = sigmoid(H1 @ W2^T + b2) ----------------
__global__ void k_gemm2(const float* __restrict__ H1, const float* __restrict__ W2,
                        const float* __restrict__ b2, float* __restrict__ out){
  int rb = blockIdx.x, kk = threadIdx.x;         // rb = b*16+h, kk in [0,64)
  const float* hr = H1 + (size_t)rb*HID;
  const float* wr = W2 + (size_t)kk*HID;
  float acc = b2[kk];
  #pragma unroll 4
  for (int o = 0; o < HID; ++o) acc = fmaf(hr[o], wr[o], acc);
  out[rb*KN + kk] = 1.f/(1.f + __expf(-acc));
}

extern "C" void kernel_launch(void* const* d_in, const int* in_sizes, int n_in,
                              void* d_out, int out_size, void* d_ws, size_t ws_size,
                              hipStream_t stream){
  const float* q   = (const float*)d_in[0];
  const float* k   = (const float*)d_in[1];
  const float* v   = (const float*)d_in[2];
  const float* Wq  = (const float*)d_in[3];
  const float* bq  = (const float*)d_in[4];
  const float* Wk  = (const float*)d_in[5];
  const float* bk  = (const float*)d_in[6];
  const float* Wv  = (const float*)d_in[7];
  const float* bv  = (const float*)d_in[8];
  const float* gq  = (const float*)d_in[9];
  const float* gk  = (const float*)d_in[10];
  const float* gv  = (const float*)d_in[11];
  const float* g1  = (const float*)d_in[12];
  const float* beq = (const float*)d_in[13];
  const float* bek = (const float*)d_in[14];
  const float* bev = (const float*)d_in[15];
  const float* be1 = (const float*)d_in[16];
  const float* W1  = (const float*)d_in[17];
  const float* b1  = (const float*)d_in[18];
  const float* W2  = (const float*)d_in[19];
  const float* b2  = (const float*)d_in[20];

  float* ws = (float*)d_ws;
  // float-offset workspace map (X bf16 reuses dead projection region at byte 0)
  const size_t OFF_PQ = 0,        OFF_PK = 1015808, OFF_PV = 2031616;
  const size_t OFF_RS = 3047424,  OFF_HP = 3048168;
  const size_t OFF_O  = 4194304;                    // above X's bf16 footprint
  const size_t OFF_OST = 12582912, OFF_OP = 12582944, OFF_H1 = 12582976;
  __bf16* X = (__bf16*)d_ws;

  k_init<<<1, 256, 0, stream>>>(ws + OFF_RS, ws + OFF_OST);

  dim3 gproj(FN, BS);
  k_proj<<<gproj, 256, 0, stream>>>(q, Wq, bq, ws + OFF_PQ, ws + OFF_RS);
  k_proj<<<gproj, 256, 0, stream>>>(k, Wk, bk, ws + OFF_PK, ws + OFF_RS + 248);
  k_proj<<<gproj, 256, 0, stream>>>(v, Wv, bv, ws + OFF_PV, ws + OFF_RS + 496);

  k_bnhead<<<1, 64, 0, stream>>>(gq, gk, gv, beq, bek, bev, ws + OFF_RS, ws + OFF_HP);

  size_t lds = (size_t)(SL*KN*2*2 + 8*512*2);       // Kn + Vn + stage = 73728 B (<320KB/WGP)
  k_attn<<<dim3(HEADS, BS), 256, lds, stream>>>(ws + OFF_PQ, ws + OFF_PK, ws + OFF_PV,
                                                ws + OFF_HP, ws + OFF_O, ws + OFF_OST);

  k_bn2<<<1, 32, 0, stream>>>(g1, be1, ws + OFF_OST, ws + OFF_OP);
  k_normo<<<32768, 256, 0, stream>>>(ws + OFF_O, ws + OFF_OP, X);

  // 65536 B dynamic LDS = two 64x128 fp32 W1 tiles (TDM double buffer)
  k_gemm1<<<dim3((HID + 63)/64, ROWS/128), 256, 65536, stream>>>(X, W1, b1, ws + OFF_H1);
  k_gemm2<<<ROWS, KN, 0, stream>>>(ws + OFF_H1, W2, b2, (float*)d_out);
}
